// GRU2_50225347559552
// MI455X (gfx1250) — matrix operand order repaired and checked by reading, not compiled, not run on hardware
//
#include <hip/hip_runtime.h>

#define Bz 128
#define Tz 512
#define Dz 256
#define Hz 256
#define G3 768
#define BT (Bz * Tz)

typedef __attribute__((ext_vector_type(16))) __bf16 v16bf;
typedef __attribute__((ext_vector_type(8)))  float  v8f;
typedef unsigned short ushort_t;

union FragB {
    uint4    u4[2];
    ushort_t us[16];
    v16bf    v;
};

static __device__ __forceinline__ ushort_t f32_to_bf16(float f) {
    unsigned u = __float_as_uint(f);
    unsigned r = u + 0x7FFFu + ((u >> 16) & 1u);   // round-to-nearest-even
    return (ushort_t)(r >> 16);
}

static __device__ __forceinline__ v8f vzero8() {
    v8f z;
#pragma unroll
    for (int i = 0; i < 8; ++i) z[i] = 0.f;
    return z;
}

static __device__ __forceinline__ float fast_tanh(float x) {
#if __has_builtin(__builtin_amdgcn_tanhf)
    return __builtin_amdgcn_tanhf(x);
#elif __has_builtin(__builtin_amdgcn_tanh_f32)
    return __builtin_amdgcn_tanh_f32(x);
#else
    return tanhf(x);
#endif
}

static __device__ __forceinline__ float fast_sigmoid(float x) {
    return __builtin_amdgcn_rcpf(1.f + __expf(-x));
}

// ---------------------------------------------------------------------------
// f32 -> bf16 elementwise conversion
// ---------------------------------------------------------------------------
__global__ void cvt_bf16(const float* __restrict__ in, ushort_t* __restrict__ out, int n) {
    int i = blockIdx.x * blockDim.x + threadIdx.x;
    if (i < n) out[i] = f32_to_bf16(in[i]);
}

// ---------------------------------------------------------------------------
// Phase 1: Xg[t][b][c] = x[b,t,:] @ w_ih.T + b_ih   (per direction)
// One wave computes a 16x64 output tile; K = 256 in 8 steps of 32.
// Throughput kernel: ~98K waves hide per-wave load latency.
// ---------------------------------------------------------------------------
__global__ __launch_bounds__(128) void xgate_gemm(
    const ushort_t* __restrict__ xbf,   // (BT, D) bf16
    const ushort_t* __restrict__ wbf,   // (3H, D) bf16
    const float*    __restrict__ bih,   // (3H)
    float*          __restrict__ Xg) {  // (T, B, 3H) f32 (dir-adjusted ptr)
    const int lane   = threadIdx.x & 31;
    const int wave   = threadIdx.x >> 5;
    const int n      = lane & 15;
    const int hi     = lane >> 4;
    const int mtile  = blockIdx.x;
    const int ntile0 = (blockIdx.y * 4 + wave) * 4;

    const ushort_t* aprow = xbf + (size_t)(mtile * 16 + n) * Dz + hi * 16;
    const ushort_t* bprow[4];
#pragma unroll
    for (int nt = 0; nt < 4; ++nt)
        bprow[nt] = wbf + (size_t)((ntile0 + nt) * 16 + n) * Dz + hi * 16;

    v8f acc[4];
#pragma unroll
    for (int nt = 0; nt < 4; ++nt) acc[nt] = vzero8();

    FragB a[2], b[2][4];
    a[0].u4[0] = *(const uint4*)(aprow);
    a[0].u4[1] = *(const uint4*)(aprow + 8);
#pragma unroll
    for (int nt = 0; nt < 4; ++nt) {
        b[0][nt].u4[0] = *(const uint4*)(bprow[nt]);
        b[0][nt].u4[1] = *(const uint4*)(bprow[nt] + 8);
    }
#pragma unroll
    for (int kk = 0; kk < 8; ++kk) {
        const int cu = kk & 1, nx = cu ^ 1;
        if (kk < 7) {   // prefetch next K-slice into the other buffer
            const int ko = (kk + 1) * 32;
            a[nx].u4[0] = *(const uint4*)(aprow + ko);
            a[nx].u4[1] = *(const uint4*)(aprow + ko + 8);
#pragma unroll
            for (int nt = 0; nt < 4; ++nt) {
                b[nx][nt].u4[0] = *(const uint4*)(bprow[nt] + ko);
                b[nx][nt].u4[1] = *(const uint4*)(bprow[nt] + ko + 8);
            }
        }
#pragma unroll
        for (int nt = 0; nt < 4; ++nt)
            acc[nt] = __builtin_amdgcn_wmma_f32_16x16x32_bf16(
                false, a[cu].v, false, b[cu][nt].v, (short)0, acc[nt], false, false);
    }

#pragma unroll
    for (int nt = 0; nt < 4; ++nt) {
        const int   col  = (ntile0 + nt) * 16 + n;
        const float bias = bih[col];
#pragma unroll
        for (int v = 0; v < 8; ++v) {
            const int row = mtile * 16 + v + hi * 8;   // row = b*T + t
            const int bb  = row >> 9;                  // / T
            const int tt  = row & (Tz - 1);
            Xg[((size_t)tt * Bz + bb) * G3 + col] = acc[nt][v] + bias;
        }
    }
}

// ---------------------------------------------------------------------------
// Phase 2: sequential GRU scan. grid = (8 batch tiles, 2 dirs), 512 threads.
// Wave w owns hidden-column tile w (16 cols) => gate tiles {w, 16+w, 32+w}.
// w_hh fragments live in registers; h tile double-buffered in LDS (bf16).
// Split-K accumulators (2 per gate) double the independent WMMA chains.
// ---------------------------------------------------------------------------
#define LDS_STRIDE 264   // 256 + 8 pad: row stride = 132 DWORDs => bank+4/row

__global__ __launch_bounds__(512) void gru_scan(
    const float* __restrict__ whh_fw, const float* __restrict__ whh_bw,
    const float* __restrict__ bhh_fw, const float* __restrict__ bhh_bw,
    const float* __restrict__ XgBase, float* __restrict__ out) {
    const int btile = blockIdx.x;
    const int dir   = blockIdx.y;
    const int lane  = threadIdx.x & 31;
    const int wave  = threadIdx.x >> 5;   // 0..15 = H column tile
    const int n     = lane & 15;
    const int hi    = lane >> 4;

    const float* whh = dir ? whh_bw : whh_fw;
    const float* bhh = dir ? bhh_bw : bhh_fw;
    const float* Xg  = XgBase + (size_t)dir * Tz * Bz * G3;

    __shared__ __align__(16) ushort_t hlds[2][16][LDS_STRIDE];

    // Preload w_hh B-fragments (bf16) into registers: 3 gates x 8 K-tiles.
    v16bf Bf[3][8];
    float bh[3];
#pragma unroll
    for (int g = 0; g < 3; ++g) {
        const int c = g * Hz + wave * 16 + n;   // w_hh row = gate column
        bh[g] = bhh[c];
#pragma unroll
        for (int kk = 0; kk < 8; ++kk) {
            FragB b;
            const float* wp = whh + (size_t)c * Hz + kk * 32 + hi * 16;
#pragma unroll
            for (int e = 0; e < 16; ++e) b.us[e] = f32_to_bf16(wp[e]);
            Bf[g][kk] = b.v;
        }
    }

    for (int i = threadIdx.x; i < 2 * 16 * LDS_STRIDE; i += 512)
        ((ushort_t*)hlds)[i] = 0;
    v8f hp = vzero8();   // this wave's slice of h (rows v+hi*8, col = colH)
    __syncthreads();

    const int colH  = wave * 16 + n;
    const int brow0 = btile * 16 + hi * 8;

    for (int s = 0; s < Tz; ++s) {
        const int t   = dir ? (Tz - 1 - s) : s;
        const int cur = s & 1;
        const int nxt = cur ^ 1;

        // Issue this step's Xg loads first: independent of LDS h / WMMA,
        // they complete under the 24-WMMA accumulation below.
        float xg[3][8];
#pragma unroll
        for (int v = 0; v < 8; ++v) {
            const size_t xb = ((size_t)t * Bz + (brow0 + v)) * G3 + colH;
            xg[0][v] = Xg[xb];
            xg[1][v] = Xg[xb + Hz];
            xg[2][v] = Xg[xb + 2 * Hz];
        }
        // Prefetch next step's Xg cachelines (global_prefetch_b8).
        if (s + 1 < Tz) {
            const int tn = dir ? (t - 1) : (t + 1);
#pragma unroll
            for (int v = 0; v < 8; ++v) {
                const size_t xb = ((size_t)tn * Bz + (brow0 + v)) * G3 + colH;
                __builtin_prefetch(&Xg[xb], 0, 3);
                __builtin_prefetch(&Xg[xb + Hz], 0, 3);
                __builtin_prefetch(&Xg[xb + 2 * Hz], 0, 3);
            }
        }

        // Pull all 8 A-fragments out of LDS as one clause of ds_load_b128.
        FragB a[8];
#pragma unroll
        for (int kk = 0; kk < 8; ++kk) {
            const ushort_t* ap = &hlds[cur][n][kk * 32 + hi * 16];
            a[kk].u4[0] = *(const uint4*)(ap);
            a[kk].u4[1] = *(const uint4*)(ap + 8);
        }

        // Split-K: 6 independent accumulation chains of depth 4.
        v8f accL[3], accH[3];
#pragma unroll
        for (int g = 0; g < 3; ++g) { accL[g] = vzero8(); accH[g] = vzero8(); }
#pragma unroll
        for (int kk = 0; kk < 4; ++kk) {
#pragma unroll
            for (int g = 0; g < 3; ++g) {
                accL[g] = __builtin_amdgcn_wmma_f32_16x16x32_bf16(
                    false, a[kk].v, false, Bf[g][kk], (short)0, accL[g], false, false);
                accH[g] = __builtin_amdgcn_wmma_f32_16x16x32_bf16(
                    false, a[kk + 4].v, false, Bf[g][kk + 4], (short)0, accH[g], false, false);
            }
        }

#pragma unroll
        for (int v = 0; v < 8; ++v) {
            const int   rowM = v + hi * 8;
            const int   b    = btile * 16 + rowM;
            const float hr = (accL[0][v] + accH[0][v]) + bh[0];
            const float hz = (accL[1][v] + accH[1][v]) + bh[1];
            const float hn = (accL[2][v] + accH[2][v]) + bh[2];
            const float r  = fast_sigmoid(xg[0][v] + hr);
            const float z  = fast_sigmoid(xg[1][v] + hz);
            const float nn = fast_tanh(xg[2][v] + r * hn);
            const float hnew = (1.f - z) * nn + z * hp[v];
            hp[v] = hnew;
            hlds[nxt][rowM][colH] = f32_to_bf16(hnew);
            out[((size_t)b * Tz + t) * (2 * Hz) + dir * Hz + colH] = hnew;
            if (s == Tz - 1)
                out[(size_t)Bz * Tz * 2 * Hz + (size_t)dir * Bz * Hz +
                    (size_t)b * Hz + colH] = hnew;
        }
        __syncthreads();
    }
}

// ---------------------------------------------------------------------------
extern "C" void kernel_launch(void* const* d_in, const int* in_sizes, int n_in,
                              void* d_out, int out_size, void* d_ws, size_t ws_size,
                              hipStream_t stream) {
    const float* x       = (const float*)d_in[0];
    const float* w_ih_fw = (const float*)d_in[1];
    const float* w_hh_fw = (const float*)d_in[2];
    const float* b_ih_fw = (const float*)d_in[3];
    const float* b_hh_fw = (const float*)d_in[4];
    const float* w_ih_bw = (const float*)d_in[5];
    const float* w_hh_bw = (const float*)d_in[6];
    const float* b_ih_bw = (const float*)d_in[7];
    const float* b_hh_bw = (const float*)d_in[8];
    float* out = (float*)d_out;

    char* ws = (char*)d_ws;
    const size_t XBF_BYTES = (size_t)BT * Dz * 2;   // 32 MB: x as bf16
    const size_t WIH_BYTES = (size_t)G3 * Dz * 2;   // 384 KB per dir
    ushort_t* xbf   = (ushort_t*)ws;
    ushort_t* wbf_f = (ushort_t*)(ws + XBF_BYTES);
    ushort_t* wbf_b = (ushort_t*)(ws + XBF_BYTES + WIH_BYTES);
    float*    Xg    = (float*)(ws + XBF_BYTES + 2 * WIH_BYTES);   // 2 x (T,B,3H) f32
    const size_t XG_DIR = (size_t)Tz * Bz * G3;

    cvt_bf16<<<(BT * Dz) / 256, 256, 0, stream>>>(x, xbf, BT * Dz);
    cvt_bf16<<<(G3 * Dz) / 256, 256, 0, stream>>>(w_ih_fw, wbf_f, G3 * Dz);
    cvt_bf16<<<(G3 * Dz) / 256, 256, 0, stream>>>(w_ih_bw, wbf_b, G3 * Dz);

    dim3 gg(BT / 16, 3, 1);   // 4096 M-tiles x 3 N-groups (4 waves * 4 tiles each)
    xgate_gemm<<<gg, 128, 0, stream>>>(xbf, wbf_f, b_ih_fw, Xg);
    xgate_gemm<<<gg, 128, 0, stream>>>(xbf, wbf_b, b_ih_bw, Xg + XG_DIR);

    gru_scan<<<dim3(Bz / 16, 2, 1), 512, 0, stream>>>(
        w_hh_fw, w_hh_bw, b_hh_fw, b_hh_bw, Xg, out);
}
